// CConvExtractor_16913581211931
// MI455X (gfx1250) — matrix operand order repaired and checked by reading, not compiled, not run on hardware
//
#include <hip/hip_runtime.h>
#include <math.h>

// ---------------- types ----------------
typedef __attribute__((ext_vector_type(16))) __bf16        v16bf;
typedef __attribute__((ext_vector_type(8)))  __bf16        v8bf;
typedef __attribute__((ext_vector_type(8)))  float         v8f;
typedef __attribute__((ext_vector_type(4)))  unsigned int  v4u;

struct U32x8 { v4u lo, hi; };

static __device__ inline v16bf make16(v4u lo, v4u hi) {
  U32x8 u; u.lo = lo; u.hi = hi;
  return __builtin_bit_cast(v16bf, u);
}

// float -> bf16 via native hardware convert (v_cvt_pk_bf16_f32 path)
static __device__ inline unsigned short f2bf(float f) {
  __bf16 h = (__bf16)f;
  return __builtin_bit_cast(unsigned short, h);
}

// ---------------- problem constants ----------------
#define BQ    8
#define HH    256
#define WW    256
#define CC    32
#define NN    4096
#define CIN   875           // 25 * 35
#define KPAD  896           // CIN padded to multiple of 32
#define HID   256
#define OUTC  64
#define TM    32            // queries per workgroup (2 WMMA M-tiles)
#define XSTR  904           // LDS x-row stride (halfwords): bank-safe, 16B aligned
#define HSTR  264           // LDS h-row stride (halfwords): bank-safe, 16B aligned

// K permutation: reference k = 35*p + c  (p=patch pixel 0..24, c=channel 0..34)
//   c < 32 (features) -> kp = 32*p + c          (pixel-major feature block, 0..799)
//   c >= 32 (metrics) -> kp = 800 + 3*p + (c-32) (metric block, 800..874)
// GEMM over K is permutation-invariant as long as W1t uses the same layout.

// ---------------- weight pre-pack kernels ----------------
// W1 [875,256] f32 -> W1t bf16 [256][896] (N-major, K permuted+padded)
__global__ __launch_bounds__(256) void pack_w1(const float* __restrict__ W1,
                                               unsigned short* __restrict__ W1t) {
  int idx = blockIdx.x * 256 + threadIdx.x;   // 896 blocks -> 229376 = 256*896
  int n = idx / KPAD;
  int k = idx - n * KPAD;                     // source k (row of W1)
  int kp;
  float v;
  if (k < CIN) {
    int p = k / 35;
    int c = k - p * 35;
    kp = (c < 32) ? (p * 32 + c) : (800 + p * 3 + (c - 32));
    v = W1[k * HID + n];
  } else {
    kp = k;                                   // pad region maps to itself
    v = 0.0f;
  }
  W1t[n * KPAD + kp] = f2bf(v);
}

// W2 [256,64] f32 -> W2t bf16 [64][256]
__global__ __launch_bounds__(256) void pack_w2(const float* __restrict__ W2,
                                               unsigned short* __restrict__ W2t) {
  int idx = blockIdx.x * 256 + threadIdx.x;   // 64 blocks -> 16384 = 64*256
  int n = idx / HID;
  int k = idx - n * HID;
  W2t[n * HID + k] = f2bf(W2[k * OUTC + n]);
}

// ---------------- fused gather + MLP kernel ----------------
__global__ __launch_bounds__(256) void fused_mlp(
    const float* __restrict__ features,     // [B,H,W,C]
    const float* __restrict__ pos,          // [B,N,2]
    const float* __restrict__ b1,           // [256]
    const float* __restrict__ b2,           // [64]
    const unsigned short* __restrict__ W1t, // bf16 [256][896] (K permuted)
    const unsigned short* __restrict__ W2t, // bf16 [64][256]
    float* __restrict__ out)                // [B*N, 64]
{
  __shared__ unsigned short xs[TM * XSTR];  // bf16 x-tile  [32][904]  (57856 B)
  __shared__ unsigned short hs[TM * HSTR];  // bf16 h-tile  [32][264]  (16896 B)

  const int t = threadIdx.x;
  const int q0 = blockIdx.x * TM;

  // ---------- Phase 1: gather patches + metrics into LDS (bf16, permuted K) ----------
  {
    const int qi = t >> 3;       // 0..31 : query within tile
    const int lg = t & 7;        // 0..7  : worker within query
    const int q  = q0 + qi;
    const int b  = q >> 12;      // q / 4096
    const float p0 = pos[q * 2 + 0];
    const float p1 = pos[q * 2 + 1];
    const float cxf = floorf(p0 * 255.0f);
    const float cyf = floorf(p1 * 255.0f);
    const int   cx = (int)cxf, cy = (int)cyf;
    const float cnx = (cxf + 0.5f) * (1.0f / 256.0f);
    const float cny = (cyf + 0.5f) * (1.0f / 256.0f);
    unsigned short* row = &xs[qi * XSTR];
    const float* fbase = features + (size_t)b * (HH * WW * CC);

    for (int p = lg; p < 25; p += 8) {
      const int dyi = p / 5, dxi = p - dyi * 5;
      const int dx = dxi - 2, dy = dyi - 2;
      const int x = cx + dx, y = cy + dy;
      const bool inb = ((unsigned)x < (unsigned)WW) & ((unsigned)y < (unsigned)HH);
      unsigned short* dstf = row + p * 32;          // feature block, 64B aligned
      if (inb) {
        const float4* fp = (const float4*)(fbase + ((size_t)(y * WW + x)) * CC);
        #pragma unroll
        for (int c8 = 0; c8 < 4; ++c8) {            // 8 channels per iteration
          float4 va = fp[c8 * 2 + 0];
          float4 vb = fp[c8 * 2 + 1];
          v8bf h;
          h[0] = (__bf16)va.x; h[1] = (__bf16)va.y;
          h[2] = (__bf16)va.z; h[3] = (__bf16)va.w;
          h[4] = (__bf16)vb.x; h[5] = (__bf16)vb.y;
          h[6] = (__bf16)vb.z; h[7] = (__bf16)vb.w;
          *(v4u*)(dstf + c8 * 8) = __builtin_bit_cast(v4u, h);  // ds_store_b128
        }
      } else {
        v4u z = {0u, 0u, 0u, 0u};
        #pragma unroll
        for (int c8 = 0; c8 < 4; ++c8) *(v4u*)(dstf + c8 * 8) = z;
      }
      const float d0 = (float)dx * (1.0f / 256.0f) + cnx - p0;
      const float d1 = (float)dy * (1.0f / 256.0f) + cny - p1;
      const float L  = sqrtf(d0 * d0 + d1 * d1);
      const float iv = 1.0f / (L + 1e-8f);
      unsigned short* dstm = row + 800 + p * 3;     // metric block
      dstm[0] = f2bf(d0 * iv);
      dstm[1] = f2bf(d1 * iv);
      dstm[2] = f2bf(L);
    }
    if (lg == 7) {               // zero K padding 875..903
      for (int k = CIN; k < XSTR; ++k) row[k] = 0;
    }
  }
  __syncthreads();

  // ---------- Phase 2: GEMM1  x[32,896] @ W1t -> h[32,256], relu ----------
  const int wave  = t >> 5;
  const int lane  = t & 31;
  const int lh    = lane >> 4;    // 0 | 1 (lane half)
  const int l16   = lane & 15;
  const int koffA = lh ? 8 : 0;   // 16-bit A-matrix lane-half K offset
  const int koffB = lh ? 8 : 0;   // 16-bit B-matrix lane-half K offset

  {
    const int nt0 = wave * 2, nt1 = wave * 2 + 1;
    v8f acc00 = {}; v8f acc01 = {};   // (mt=0, nt0/nt1)
    v8f acc10 = {}; v8f acc11 = {};   // (mt=1, nt0/nt1)
    const unsigned short* A0 = &xs[l16 * XSTR];
    const unsigned short* A1 = &xs[(16 + l16) * XSTR];
    const unsigned short* B0 = W1t + (size_t)(nt0 * 16 + l16) * KPAD;
    const unsigned short* B1 = W1t + (size_t)(nt1 * 16 + l16) * KPAD;

    for (int kt = 0; kt < 28; ++kt) {
      const int kb = kt * 32;
      v16bf a0 = make16(*(const v4u*)(A0 + kb + koffA),
                        *(const v4u*)(A0 + kb + koffA + 16));
      v16bf a1 = make16(*(const v4u*)(A1 + kb + koffA),
                        *(const v4u*)(A1 + kb + koffA + 16));
      v16bf bb0 = make16(*(const v4u*)(B0 + kb + koffB),
                         *(const v4u*)(B0 + kb + koffB + 16));
      v16bf bb1 = make16(*(const v4u*)(B1 + kb + koffB),
                         *(const v4u*)(B1 + kb + koffB + 16));
      acc00 = __builtin_amdgcn_wmma_f32_16x16x32_bf16(false, a0, false, bb0,
                                                      (short)0, acc00, false, false);
      acc10 = __builtin_amdgcn_wmma_f32_16x16x32_bf16(false, a1, false, bb0,
                                                      (short)0, acc10, false, false);
      acc01 = __builtin_amdgcn_wmma_f32_16x16x32_bf16(false, a0, false, bb1,
                                                      (short)0, acc01, false, false);
      acc11 = __builtin_amdgcn_wmma_f32_16x16x32_bf16(false, a1, false, bb1,
                                                      (short)0, acc11, false, false);
    }

    const float bias0 = b1[nt0 * 16 + l16];
    const float bias1 = b1[nt1 * 16 + l16];
    const int Mb = lh * 8;
    #pragma unroll
    for (int r = 0; r < 8; ++r) {
      const int M0 = Mb + r;         // mt = 0
      const int M1 = 16 + Mb + r;    // mt = 1
      float h00 = acc00[r] + bias0; h00 = h00 > 0.0f ? h00 : 0.0f;
      float h01 = acc01[r] + bias1; h01 = h01 > 0.0f ? h01 : 0.0f;
      float h10 = acc10[r] + bias0; h10 = h10 > 0.0f ? h10 : 0.0f;
      float h11 = acc11[r] + bias1; h11 = h11 > 0.0f ? h11 : 0.0f;
      hs[M0 * HSTR + nt0 * 16 + l16] = f2bf(h00);
      hs[M0 * HSTR + nt1 * 16 + l16] = f2bf(h01);
      hs[M1 * HSTR + nt0 * 16 + l16] = f2bf(h10);
      hs[M1 * HSTR + nt1 * 16 + l16] = f2bf(h11);
    }
  }
  __syncthreads();

  // ---------- Phase 3: GEMM2  h[32,256] @ W2t -> out[32,64] ----------
  {
    const int mt = wave >> 2;       // 0..1
    const int nt = wave & 3;        // 0..3
    v8f acc = {};
    const unsigned short* Ah = &hs[(mt * 16 + l16) * HSTR];
    const unsigned short* Bw = W2t + (size_t)(nt * 16 + l16) * HID;

    #pragma unroll
    for (int kt = 0; kt < 8; ++kt) {
      const int kb = kt * 32;
      v16bf a = make16(*(const v4u*)(Ah + kb + koffA),
                       *(const v4u*)(Ah + kb + koffA + 16));
      v16bf b = make16(*(const v4u*)(Bw + kb + koffB),
                       *(const v4u*)(Bw + kb + koffB + 16));
      acc = __builtin_amdgcn_wmma_f32_16x16x32_bf16(false, a, false, b,
                                                    (short)0, acc, false, false);
    }

    const float bias = b2[nt * 16 + l16];
    const int Mb = mt * 16 + lh * 8;
    #pragma unroll
    for (int r = 0; r < 8; ++r) {
      out[(size_t)(q0 + Mb + r) * OUTC + nt * 16 + l16] = acc[r] + bias;
    }
  }
}

// ---------------- host launcher ----------------
extern "C" void kernel_launch(void* const* d_in, const int* in_sizes, int n_in,
                              void* d_out, int out_size, void* d_ws, size_t ws_size,
                              hipStream_t stream) {
  const float* features = (const float*)d_in[0];
  const float* pos      = (const float*)d_in[1];
  const float* W1       = (const float*)d_in[2];
  const float* b1       = (const float*)d_in[3];
  const float* W2       = (const float*)d_in[4];
  const float* b2       = (const float*)d_in[5];

  unsigned short* W1t = (unsigned short*)d_ws;                                   // 458752 B
  unsigned short* W2t = (unsigned short*)((char*)d_ws + (size_t)HID * KPAD * 2); // +32768 B

  pack_w1<<<KPAD, 256, 0, stream>>>(W1, W1t);       // 896 blocks
  pack_w2<<<OUTC, 256, 0, stream>>>(W2, W2t);       // 64 blocks

  const int total_q = BQ * NN;                      // 32768
  fused_mlp<<<total_q / TM, 256, 0, stream>>>(features, pos, b1, b2, W1t, W2t,
                                              (float*)d_out);
}
